// SIN0_1236950582134
// MI455X (gfx1250) — matrix operand order repaired and checked by reading, not compiled, not run on hardware
//
#include <hip/hip_runtime.h>
#include <hip/hip_bf16.h>

typedef __attribute__((ext_vector_type(16))) _Float16 v16h;
typedef __attribute__((ext_vector_type(8)))  _Float16 v8h;
typedef __attribute__((ext_vector_type(8)))  float    v8f;
typedef __attribute__((ext_vector_type(4)))  float    v4f;
typedef unsigned int u32x4 __attribute__((ext_vector_type(4)));
typedef int          i32x4 __attribute__((ext_vector_type(4)));
typedef int          i32x8 __attribute__((ext_vector_type(8)));

#define SIN_WAVES 8

__device__ __forceinline__ v16h sin_cat8(v8h a, v8h b) {
    return __builtin_shufflevector(a, b, 0,1,2,3,4,5,6,7,8,9,10,11,12,13,14,15);
}

// -------------------------------------------------------------------------
// Tensor Data Mover: DMA a rows x cols f32 tile (row-major, contiguous)
// from global memory into LDS.  D# layout per CDNA5 ISA ch.8:
//   group0: count=1 | lds_addr[63:32] | global_addr[120:64] | type=2
//   group1: data_size=4B, tensor_dim0=cols, tensor_dim1=rows,
//           tile_dim0=cols, tile_dim1=rows, tensor_dim0_stride=cols
// Generic LDS pointer truncated to 32 bits == LDS byte offset (ISA 10.2).
// -------------------------------------------------------------------------
__device__ __forceinline__ void sin_tdm_load_f32(void* ldsDst, const void* gsrc,
                                                 unsigned rows, unsigned cols)
{
    const unsigned lds = (unsigned)(size_t)ldsDst;
    const unsigned long long ga = (unsigned long long)(size_t)gsrc;
    u32x4 g0;
    g0[0] = 1u;                                           // count=1 (valid user D#)
    g0[1] = lds;                                          // lds_addr
    g0[2] = (unsigned)(ga & 0xFFFFFFFFu);                 // global_addr[31:0]
    g0[3] = (unsigned)((ga >> 32) & 0x01FFFFFFu) | (2u << 30); // addr[56:32] | type=2
    i32x8 g1;
    g1[0] = (2 << 16);                                    // data_size = 4 bytes
    g1[1] = (int)((cols & 0xFFFFu) << 16);                // tensor_dim0[15:0]
    g1[2] = (int)(((cols >> 16) & 0xFFFFu) | ((rows & 0xFFFFu) << 16)); // dim0 hi | dim1 lo
    g1[3] = (int)(((rows >> 16) & 0xFFFFu) | ((cols & 0xFFFFu) << 16)); // dim1 hi | tile_dim0
    g1[4] = (int)(rows & 0xFFFFu);                        // tile_dim1 (tile_dim2 = 0)
    g1[5] = (int)cols;                                    // tensor_dim0_stride[31:0]
    g1[6] = 0;
    g1[7] = 0;
    const i32x4 z4 = {0, 0, 0, 0};
#if __has_include(<hip/amd_detail/amd_gfx1250_TDM.h>)
    const i32x8 z8 = {0, 0, 0, 0, 0, 0, 0, 0};
    __builtin_amdgcn_tensor_load_to_lds(g0, g1, z4, z4, z8, 0);
#else
    __builtin_amdgcn_tensor_load_to_lds(g0, g1, z4, z4, 0);
#endif
}

// -------------------------------------------------------------------------
// Message kernel: per wave, a 16-edge tile.
//   m = relu(concat(x_src[src], attr[cob]) @ W + b)          (WMMA f16, K=128)
//   atomicAdd agg[dst] += m ; deg[dst] += 1 ; column sum/sumsq -> stats
// BatchNorm applied later as per-column affine via (s,t) + deg.
// Fully branchless tail handling: clamped indices + cndmask-zeroed values.
// -------------------------------------------------------------------------
__global__ __launch_bounds__(256) void sin_msg_kernel(
    const float* __restrict__ xsrc, const float* __restrict__ xattr,
    const int* __restrict__ srcIdx, const int* __restrict__ dstIdx,
    const int* __restrict__ shIdx, int E,
    const float* __restrict__ Wg, const float* __restrict__ bg,
    float* __restrict__ agg, float* __restrict__ deg,
    float* __restrict__ gSum, float* __restrict__ gSq)
{
    __shared__ float    sWstage[128 * 64];            // 32 KB TDM landing pad
    __shared__ _Float16 sW[128 * 64];                 // 16 KB f16 weights
    __shared__ _Float16 sA[SIN_WAVES][16 * 136];      // per-wave A staging (pad 8)
    __shared__ int sSrc[SIN_WAVES][16];
    __shared__ int sCob[SIN_WAVES][16];
    __shared__ int sDst[SIN_WAVES][16];
    __shared__ float sSum[64], sSq[64];

    const int tid  = threadIdx.x;
    const int wave = tid >> 5;
    const int lane = tid & 31;
    const int l15  = lane & 15;
    const int hi   = lane >> 4;

    if (wave == 0) {
        sin_tdm_load_f32(sWstage, Wg, 128, 64);       // async DMA, TENSORcnt++
        __builtin_amdgcn_s_wait_tensorcnt(0);
    }
    if (tid < 64) { sSum[tid] = 0.f; sSq[tid] = 0.f; }
    __syncthreads();
    for (int i = tid; i < 128 * 64; i += 256) sW[i] = (_Float16)sWstage[i];
    __syncthreads();

    // persistent B fragments: lane holds column n = nt*16+l15,
    // K = kt*32 + hi*16 + 0..15
    v16h bf[4][4];
#pragma unroll
    for (int nt = 0; nt < 4; ++nt) {
        const int n = nt * 16 + l15;
#pragma unroll
        for (int kt = 0; kt < 4; ++kt) {
            const int k0 = kt * 32 + hi * 16;
            v16h b;
#pragma unroll
            for (int i = 0; i < 16; ++i) b[i] = sW[(k0 + i) * 64 + n];
            bf[nt][kt] = b;
        }
    }

    float bias[4];
#pragma unroll
    for (int nt = 0; nt < 4; ++nt) bias[nt] = bg[nt * 16 + l15];

    float ls[4] = {0.f, 0.f, 0.f, 0.f};
    float lq[4] = {0.f, 0.f, 0.f, 0.f};

    const int tiles  = (E + 15) >> 4;
    const int stride = gridDim.x * SIN_WAVES;
    for (int tile = blockIdx.x * SIN_WAVES + wave; tile < tiles; tile += stride) {
        const int e0 = tile * 16;
        const int ePf = (e0 + stride * 16 < E) ? e0 + stride * 16 : e0;
        __builtin_prefetch(srcIdx + ePf, 0, 1);       // global_prefetch_b8
        __builtin_prefetch(shIdx + ePf, 0, 1);

        // stage per-row edge indices (branchless; clamped loads, masked deg)
        if (lane < 16) {
            const int e  = e0 + lane;
            const int ec = e < E ? e : E - 1;
            const bool ok = e < E;
            const int s = srcIdx[ec];
            const int d = dstIdx[ec];
            atomicAdd(&deg[d], ok ? 1.0f : 0.0f);
            sSrc[wave][lane] = ok ? s : -1;
            sDst[wave][lane] = ok ? d : -1;
        } else {
            const int e  = e0 + lane - 16;
            const int ec = e < E ? e : E - 1;
            sCob[wave][lane - 16] = (e < E) ? shIdx[ec] : -1;
        }
        asm volatile("s_wait_dscnt 0" ::: "memory");   // wave-local LDS ordering

        // gather: iteration `row` covers one edge row; 32 lanes = 128 floats
#pragma unroll 4
        for (int row = 0; row < 16; ++row) {
            const int   src  = (lane < 16) ? sSrc[wave][row] : sCob[wave][row];
            const float* bse = (lane < 16) ? xsrc : xattr;
            const int   col  = (lane < 16) ? lane * 4 : (lane - 16) * 4;
            const int   sc   = src >= 0 ? src : 0;
            const float msk  = src >= 0 ? 1.0f : 0.0f;
            const v4f v = *(const v4f*)(bse + (size_t)sc * 64 + col);
            _Float16* dst = &sA[wave][row * 136 +
                                      ((lane < 16) ? lane * 4 : 64 + (lane - 16) * 4)];
            dst[0] = (_Float16)(v.x * msk); dst[1] = (_Float16)(v.y * msk);
            dst[2] = (_Float16)(v.z * msk); dst[3] = (_Float16)(v.w * msk);
        }
        asm volatile("s_wait_dscnt 0" ::: "memory");

        // A fragments: lane = row l15; K = hi*16 + 0..15
        v16h af[4];
#pragma unroll
        for (int kt = 0; kt < 4; ++kt) {
            const _Float16* rp = &sA[wave][l15 * 136 + kt * 32 + hi * 16];
            af[kt] = sin_cat8(*(const v8h*)rp, *(const v8h*)(rp + 8));
        }

        v8f c[4];
#pragma unroll
        for (int nt = 0; nt < 4; ++nt) {
            v8f acc = {};
#pragma unroll
            for (int kt = 0; kt < 4; ++kt)
                acc = __builtin_amdgcn_wmma_f32_16x16x32_f16(
                    false, af[kt], false, bf[nt][kt], (short)0, acc, false, false);
            c[nt] = acc;
        }

        int dsts[8];
#pragma unroll
        for (int r = 0; r < 8; ++r) dsts[r] = sDst[wave][hi * 8 + r];

        // branchless scatter: invalid rows add 0.0f to row 0
#pragma unroll
        for (int nt = 0; nt < 4; ++nt) {
#pragma unroll
            for (int r = 0; r < 8; ++r) {
                const int d  = dsts[r];
                const int dc = d >= 0 ? d : 0;
                float v = c[nt][r] + bias[nt];
                v = v > 0.f ? v : 0.f;
                v = d >= 0 ? v : 0.f;
                ls[nt] += v;
                lq[nt] += v * v;
                atomicAdd(&agg[(size_t)dc * 64 + nt * 16 + l15], v);
            }
        }
    }

#pragma unroll
    for (int nt = 0; nt < 4; ++nt) {
        atomicAdd(&sSum[nt * 16 + l15], ls[nt]);
        atomicAdd(&sSq [nt * 16 + l15], lq[nt]);
    }
    __syncthreads();
    if (tid < 64) {
        atomicAdd(&gSum[tid], sSum[tid]);
        atomicAdd(&gSq [tid], sSq [tid]);
    }
}

// -------------------------------------------------------------------------
// Update kernel: h = x + s1*agg1 + t1*deg1 (+ s2*agg2 + t2*deg2)
//   h2 = relu(relu(h@W1+b1)@W2+b2) -> hOut (+ column stats for the final BN)
// hOut must have capacity for N+16 rows (invalid tail rows dump to row N).
// -------------------------------------------------------------------------
__global__ __launch_bounds__(256) void sin_update_kernel(
    const float* __restrict__ x,
    const float* __restrict__ agg1, const float* __restrict__ deg1,
    const float* __restrict__ aff1,
    const float* __restrict__ agg2, const float* __restrict__ deg2,
    const float* __restrict__ aff2,
    int N,
    const float* __restrict__ W1g, const float* __restrict__ b1g,
    const float* __restrict__ W2g, const float* __restrict__ b2g,
    float* __restrict__ hOut, float* __restrict__ gSum, float* __restrict__ gSq)
{
    __shared__ float    sWstage[2 * 64 * 64];         // 32 KB TDM landing pad
    __shared__ _Float16 sW1[64 * 64];
    __shared__ _Float16 sW2[64 * 64];
    __shared__ _Float16 sH[SIN_WAVES][16 * 72];
    __shared__ float sSum[64], sSq[64];

    const int tid  = threadIdx.x;
    const int wave = tid >> 5;
    const int lane = tid & 31;
    const int l15  = lane & 15;
    const int hi   = lane >> 4;

    if (wave == 0) {
        sin_tdm_load_f32(sWstage,        W1g, 64, 64);
        sin_tdm_load_f32(sWstage + 4096, W2g, 64, 64);
        __builtin_amdgcn_s_wait_tensorcnt(0);
    }
    if (tid < 64) { sSum[tid] = 0.f; sSq[tid] = 0.f; }
    __syncthreads();
    for (int i = tid; i < 64 * 64; i += 256) {
        sW1[i] = (_Float16)sWstage[i];
        sW2[i] = (_Float16)sWstage[4096 + i];
    }
    __syncthreads();

    v16h bf1[4][2], bf2[4][2];
#pragma unroll
    for (int nt = 0; nt < 4; ++nt) {
        const int n = nt * 16 + l15;
#pragma unroll
        for (int kt = 0; kt < 2; ++kt) {
            const int k0 = kt * 32 + hi * 16;
            v16h a, b;
#pragma unroll
            for (int i = 0; i < 16; ++i) {
                a[i] = sW1[(k0 + i) * 64 + n];
                b[i] = sW2[(k0 + i) * 64 + n];
            }
            bf1[nt][kt] = a;
            bf2[nt][kt] = b;
        }
    }

    float bias1[4], bias2[4];
#pragma unroll
    for (int nt = 0; nt < 4; ++nt) {
        bias1[nt] = b1g[nt * 16 + l15];
        bias2[nt] = b2g[nt * 16 + l15];
    }

    // per-lane affine for the gather columns (cols = l15*4 .. l15*4+3)
    const v4f s1v = *(const v4f*)(aff1 + l15 * 4);
    const v4f t1v = *(const v4f*)(aff1 + 64 + l15 * 4);
    v4f s2v = {0.f, 0.f, 0.f, 0.f}, t2v = {0.f, 0.f, 0.f, 0.f};
    const bool has2 = (agg2 != nullptr);
    if (has2) {
        s2v = *(const v4f*)(aff2 + l15 * 4);
        t2v = *(const v4f*)(aff2 + 64 + l15 * 4);
    }

    float ls[4] = {0.f, 0.f, 0.f, 0.f};
    float lq[4] = {0.f, 0.f, 0.f, 0.f};

    const int tiles = (N + 15) >> 4;
    for (int tile = blockIdx.x * SIN_WAVES + wave; tile < tiles;
         tile += gridDim.x * SIN_WAVES) {
        const int n0 = tile * 16;

        // gather + residual + folded-BN aggregation; 2 rows per iteration
#pragma unroll 4
        for (int it = 0; it < 8; ++it) {
            const int row  = it * 2 + hi;
            const int node = n0 + row;
            const int nc   = node < N ? node : N - 1;     // clamped, always safe
            const size_t base = (size_t)nc * 64 + l15 * 4;
            const v4f xv = *(const v4f*)(x + base);
            const v4f a1 = *(const v4f*)(agg1 + base);
            const float d1 = deg1[nc];
            v4f h = xv + s1v * a1 + t1v * d1;
            if (has2) {
                const v4f a2 = *(const v4f*)(agg2 + base);
                const float d2 = deg2[nc];
                h = h + s2v * a2 + t2v * d2;
            }
            _Float16* dst = &sH[wave][row * 72 + l15 * 4];
            dst[0] = (_Float16)h.x; dst[1] = (_Float16)h.y;
            dst[2] = (_Float16)h.z; dst[3] = (_Float16)h.w;
        }
        asm volatile("s_wait_dscnt 0" ::: "memory");

        v16h af[2];
#pragma unroll
        for (int kt = 0; kt < 2; ++kt) {
            const _Float16* rp = &sH[wave][l15 * 72 + kt * 32 + hi * 16];
            af[kt] = sin_cat8(*(const v8h*)rp, *(const v8h*)(rp + 8));
        }

        // GEMM1 + relu, write h1 back into the staging tile in A layout
        v8f c[4];
#pragma unroll
        for (int nt = 0; nt < 4; ++nt) {
            v8f acc = {};
#pragma unroll
            for (int kt = 0; kt < 2; ++kt)
                acc = __builtin_amdgcn_wmma_f32_16x16x32_f16(
                    false, af[kt], false, bf1[nt][kt], (short)0, acc, false, false);
            c[nt] = acc;
        }
        asm volatile("s_wait_dscnt 0" ::: "memory");  // A frags consumed before overwrite
#pragma unroll
        for (int nt = 0; nt < 4; ++nt) {
#pragma unroll
            for (int r = 0; r < 8; ++r) {
                float v = c[nt][r] + bias1[nt];
                v = v > 0.f ? v : 0.f;
                sH[wave][(hi * 8 + r) * 72 + nt * 16 + l15] = (_Float16)v;
            }
        }
        asm volatile("s_wait_dscnt 0" ::: "memory");

#pragma unroll
        for (int kt = 0; kt < 2; ++kt) {
            const _Float16* rp = &sH[wave][l15 * 72 + kt * 32 + hi * 16];
            af[kt] = sin_cat8(*(const v8h*)rp, *(const v8h*)(rp + 8));
        }

        // GEMM2 + relu + stats + store h2 (branchless: tail rows -> dump row N)
#pragma unroll
        for (int nt = 0; nt < 4; ++nt) {
            v8f acc = {};
#pragma unroll
            for (int kt = 0; kt < 2; ++kt)
                acc = __builtin_amdgcn_wmma_f32_16x16x32_f16(
                    false, af[kt], false, bf2[nt][kt], (short)0, acc, false, false);
#pragma unroll
            for (int r = 0; r < 8; ++r) {
                const int node = n0 + hi * 8 + r;
                const int rowi = node < N ? node : N;
                float v = acc[r] + bias2[nt];
                v = v > 0.f ? v : 0.f;
                const float vm = node < N ? v : 0.f;
                ls[nt] += vm;
                lq[nt] += vm * vm;
                hOut[(size_t)rowi * 64 + nt * 16 + l15] = v;
            }
        }
    }

#pragma unroll
    for (int nt = 0; nt < 4; ++nt) {
        atomicAdd(&sSum[nt * 16 + l15], ls[nt]);
        atomicAdd(&sSq [nt * 16 + l15], lq[nt]);
    }
    __syncthreads();
    if (tid < 64) {
        atomicAdd(&gSum[tid], sSum[tid]);
        atomicAdd(&gSq [tid], sSq [tid]);
    }
}

// stats -> per-column affine (s, t): bn(v) = s*v + t
__global__ void sin_finalize_kernel(const float* __restrict__ gSum,
                                    const float* __restrict__ gSq, float cnt,
                                    const float* __restrict__ g,
                                    const float* __restrict__ bb,
                                    float* __restrict__ aff)
{
    const int c = threadIdx.x;
    if (c < 64) {
        const float mu   = gSum[c] / cnt;
        const float var  = gSq[c] / cnt - mu * mu;
        const float rstd = rsqrtf(var + 1e-5f);
        const float s    = g[c] * rstd;
        aff[c]      = s;
        aff[64 + c] = bb[c] - mu * s;
    }
}

__global__ void sin_bnapply_kernel(const float* __restrict__ h,
                                   const float* __restrict__ aff, int n64,
                                   float* __restrict__ xout)
{
    const int t = blockIdx.x * blockDim.x + threadIdx.x;
    if (t < n64) {
        const int c = t & 63;
        xout[t] = h[t] * aff[c] + aff[64 + c];
    }
}

__global__ void sin_pool_kernel(const float* __restrict__ x,
                                const int* __restrict__ batch, int n64,
                                float* __restrict__ pooled)
{
    const int t = blockIdx.x * blockDim.x + threadIdx.x;
    if (t < n64) {
        const int n = t >> 6, c = t & 63;
        atomicAdd(&pooled[(size_t)batch[n] * 64 + c], x[t]);
    }
}

__global__ __launch_bounds__(64) void sin_head_kernel(
    const float* __restrict__ pooled,
    const float* __restrict__ l1w, const float* __restrict__ l1b,
    const float* __restrict__ l2w, const float* __restrict__ l2b,
    float* __restrict__ out)
{
    __shared__ float sp[64], sh[64];
    const int b = blockIdx.x, t = threadIdx.x;
    sp[t] = pooled[(size_t)b * 64 + t];
    __syncthreads();
    float acc = l1b[t];
#pragma unroll
    for (int k = 0; k < 64; ++k) acc += sp[k] * l1w[k * 64 + t];
    sh[t] = acc > 0.f ? acc : 0.f;
    __syncthreads();
    if (t < 6) {
        float o = l2b[t];
#pragma unroll
        for (int k = 0; k < 64; ++k) o += sh[k] * l2w[k * 6 + t];
        out[b * 6 + t] = o;
    }
}

// -------------------------------------------------------------------------
extern "C" void kernel_launch(void* const* d_in, const int* in_sizes, int n_in,
                              void* d_out, int out_size, void* d_ws, size_t ws_size,
                              hipStream_t stream)
{
    (void)in_sizes; (void)n_in; (void)out_size;
    constexpr int N0 = 100000, N1 = 200000, N2 = 100000;
    constexpr int E0U = 400000, E1U = 600000, E1D = 800000, E2D = 600000;
    constexpr int B = 1024, L = 3;

    const float* x0 = (const float*)d_in[0];
    const float* x1 = (const float*)d_in[1];
    const float* x2 = (const float*)d_in[2];
    const int* ui0  = (const int*)d_in[3];
    const int* cob0 = (const int*)d_in[4];
    const int* ui1  = (const int*)d_in[5];
    const int* cob1 = (const int*)d_in[6];
    const int* di1  = (const int*)d_in[7];
    const int* bnd1 = (const int*)d_in[8];
    const int* di2  = (const int*)d_in[9];
    const int* bnd2 = (const int*)d_in[10];
    const int* bat0 = (const int*)d_in[11];
    const int* bat1 = (const int*)d_in[12];
    const int* bat2 = (const int*)d_in[13];
    const float* W_up  = (const float*)d_in[14];
    const float* b_up  = (const float*)d_in[15];
    const float* g_up  = (const float*)d_in[16];
    const float* bb_up = (const float*)d_in[17];
    const float* W_dn  = (const float*)d_in[18];
    const float* b_dn  = (const float*)d_in[19];
    const float* g_dn  = (const float*)d_in[20];
    const float* bb_dn = (const float*)d_in[21];
    const float* W1p   = (const float*)d_in[22];
    const float* b1p   = (const float*)d_in[23];
    const float* W2p   = (const float*)d_in[24];
    const float* b2p   = (const float*)d_in[25];
    const float* g_u   = (const float*)d_in[26];
    const float* bb_u  = (const float*)d_in[27];
    const float* lin1w = (const float*)d_in[28];
    const float* lin1b = (const float*)d_in[29];
    const float* lin2w = (const float*)d_in[30];
    const float* lin2b = (const float*)d_in[31];

    char* wsb = (char*)d_ws;
    size_t off = 0;
    auto alloc = [&](size_t bytes) -> float* {
        float* p = (float*)(wsb + off);
        off += (bytes + 255) & ~(size_t)255;
        return p;
    };

    float* X0 = alloc((size_t)N0 * 64 * 4);
    float* X1 = alloc((size_t)N1 * 64 * 4);
    float* X2 = alloc((size_t)N2 * 64 * 4);
    float* AGG0  = alloc((size_t)N0 * 64 * 4);
    float* AGGU1 = alloc((size_t)N1 * 64 * 4);
    float* AGGD1 = alloc((size_t)N1 * 64 * 4);
    float* AGGD2 = alloc((size_t)N2 * 64 * 4);
    float* DEG0  = alloc((size_t)N0 * 4);
    float* DEGU1 = alloc((size_t)N1 * 4);
    float* DEGD1 = alloc((size_t)N1 * 4);
    float* DEGD2 = alloc((size_t)N2 * 4);
    float* H2    = alloc((size_t)(N1 + 16) * 64 * 4);  // + dump tile
    float* POOL  = alloc((size_t)B * 64 * 4);
    float* STAT  = alloc(5 * 128 * 4);                 // sum(64)+sq(64) per set
    float* AFF   = alloc(5 * 128 * 4);                 // s(64)+t(64) per set
    if (off > ws_size) return;                         // workspace too small

    hipMemcpyAsync(X0, x0, (size_t)N0 * 64 * 4, hipMemcpyDeviceToDevice, stream);
    hipMemcpyAsync(X1, x1, (size_t)N1 * 64 * 4, hipMemcpyDeviceToDevice, stream);
    hipMemcpyAsync(X2, x2, (size_t)N2 * 64 * 4, hipMemcpyDeviceToDevice, stream);

    auto nblocks = [](int n) {
        int t = ((n + 15) / 16 + SIN_WAVES - 1) / SIN_WAVES;
        return t < 1920 ? t : 1920;
    };

    for (int l = 0; l < L; ++l) {
        hipMemsetAsync(AGG0,  0, (size_t)N0 * 64 * 4, stream);
        hipMemsetAsync(AGGU1, 0, (size_t)N1 * 64 * 4, stream);
        hipMemsetAsync(AGGD1, 0, (size_t)N1 * 64 * 4, stream);
        hipMemsetAsync(AGGD2, 0, (size_t)N2 * 64 * 4, stream);
        hipMemsetAsync(DEG0,  0, (size_t)N0 * 4, stream);
        hipMemsetAsync(DEGU1, 0, (size_t)N1 * 4, stream);
        hipMemsetAsync(DEGD1, 0, (size_t)N1 * 4, stream);
        hipMemsetAsync(DEGD2, 0, (size_t)N2 * 4, stream);
        hipMemsetAsync(STAT, 0, 4 * 128 * 4, stream);

        const int d0 = l * 3 + 0, d1 = l * 3 + 1, d2 = l * 3 + 2;

        sin_msg_kernel<<<nblocks(E0U), 256, 0, stream>>>(
            X0, X1, ui0, ui0 + E0U, cob0, E0U,
            W_up + (size_t)d0 * 8192, b_up + (size_t)d0 * 64,
            AGG0, DEG0, STAT + 0 * 128, STAT + 0 * 128 + 64);
        sin_msg_kernel<<<nblocks(E1U), 256, 0, stream>>>(
            X1, X2, ui1, ui1 + E1U, cob1, E1U,
            W_up + (size_t)d1 * 8192, b_up + (size_t)d1 * 64,
            AGGU1, DEGU1, STAT + 1 * 128, STAT + 1 * 128 + 64);
        sin_msg_kernel<<<nblocks(E1D), 256, 0, stream>>>(
            X1, X0, di1, di1 + E1D, bnd1, E1D,
            W_dn + (size_t)d1 * 8192, b_dn + (size_t)d1 * 64,
            AGGD1, DEGD1, STAT + 2 * 128, STAT + 2 * 128 + 64);
        sin_msg_kernel<<<nblocks(E2D), 256, 0, stream>>>(
            X2, X1, di2, di2 + E2D, bnd2, E2D,
            W_dn + (size_t)d2 * 8192, b_dn + (size_t)d2 * 64,
            AGGD2, DEGD2, STAT + 3 * 128, STAT + 3 * 128 + 64);

        sin_finalize_kernel<<<1, 64, 0, stream>>>(
            STAT + 0 * 128, STAT + 0 * 128 + 64, (float)E0U,
            g_up + (size_t)d0 * 64, bb_up + (size_t)d0 * 64, AFF + 0 * 128);
        sin_finalize_kernel<<<1, 64, 0, stream>>>(
            STAT + 1 * 128, STAT + 1 * 128 + 64, (float)E1U,
            g_up + (size_t)d1 * 64, bb_up + (size_t)d1 * 64, AFF + 1 * 128);
        sin_finalize_kernel<<<1, 64, 0, stream>>>(
            STAT + 2 * 128, STAT + 2 * 128 + 64, (float)E1D,
            g_dn + (size_t)d1 * 64, bb_dn + (size_t)d1 * 64, AFF + 2 * 128);
        sin_finalize_kernel<<<1, 64, 0, stream>>>(
            STAT + 3 * 128, STAT + 3 * 128 + 64, (float)E2D,
            g_dn + (size_t)d2 * 64, bb_dn + (size_t)d2 * 64, AFF + 3 * 128);

        // ---- level 0 update ----
        hipMemsetAsync(STAT + 4 * 128, 0, 128 * 4, stream);
        sin_update_kernel<<<nblocks(N0), 256, 0, stream>>>(
            X0, AGG0, DEG0, AFF + 0 * 128, nullptr, nullptr, nullptr, N0,
            W1p + (size_t)d0 * 4096, b1p + (size_t)d0 * 64,
            W2p + (size_t)d0 * 4096, b2p + (size_t)d0 * 64,
            H2, STAT + 4 * 128, STAT + 4 * 128 + 64);
        sin_finalize_kernel<<<1, 64, 0, stream>>>(
            STAT + 4 * 128, STAT + 4 * 128 + 64, (float)N0,
            g_u + (size_t)d0 * 64, bb_u + (size_t)d0 * 64, AFF + 4 * 128);
        sin_bnapply_kernel<<<(N0 * 64 + 255) / 256, 256, 0, stream>>>(
            H2, AFF + 4 * 128, N0 * 64, X0);

        // ---- level 1 update (up1 + dn1) ----
        hipMemsetAsync(STAT + 4 * 128, 0, 128 * 4, stream);
        sin_update_kernel<<<nblocks(N1), 256, 0, stream>>>(
            X1, AGGU1, DEGU1, AFF + 1 * 128, AGGD1, DEGD1, AFF + 2 * 128, N1,
            W1p + (size_t)d1 * 4096, b1p + (size_t)d1 * 64,
            W2p + (size_t)d1 * 4096, b2p + (size_t)d1 * 64,
            H2, STAT + 4 * 128, STAT + 4 * 128 + 64);
        sin_finalize_kernel<<<1, 64, 0, stream>>>(
            STAT + 4 * 128, STAT + 4 * 128 + 64, (float)N1,
            g_u + (size_t)d1 * 64, bb_u + (size_t)d1 * 64, AFF + 4 * 128);
        sin_bnapply_kernel<<<(N1 * 64 + 255) / 256, 256, 0, stream>>>(
            H2, AFF + 4 * 128, N1 * 64, X1);

        // ---- level 2 update ----
        hipMemsetAsync(STAT + 4 * 128, 0, 128 * 4, stream);
        sin_update_kernel<<<nblocks(N2), 256, 0, stream>>>(
            X2, AGGD2, DEGD2, AFF + 3 * 128, nullptr, nullptr, nullptr, N2,
            W1p + (size_t)d2 * 4096, b1p + (size_t)d2 * 64,
            W2p + (size_t)d2 * 4096, b2p + (size_t)d2 * 64,
            H2, STAT + 4 * 128, STAT + 4 * 128 + 64);
        sin_finalize_kernel<<<1, 64, 0, stream>>>(
            STAT + 4 * 128, STAT + 4 * 128 + 64, (float)N2,
            g_u + (size_t)d2 * 64, bb_u + (size_t)d2 * 64, AFF + 4 * 128);
        sin_bnapply_kernel<<<(N2 * 64 + 255) / 256, 256, 0, stream>>>(
            H2, AFF + 4 * 128, N2 * 64, X2);
    }

    hipMemsetAsync(POOL, 0, (size_t)B * 64 * 4, stream);
    sin_pool_kernel<<<(N0 * 64 + 255) / 256, 256, 0, stream>>>(X0, bat0, N0 * 64, POOL);
    sin_pool_kernel<<<(N1 * 64 + 255) / 256, 256, 0, stream>>>(X1, bat1, N1 * 64, POOL);
    sin_pool_kernel<<<(N2 * 64 + 255) / 256, 256, 0, stream>>>(X2, bat2, N2 * 64, POOL);

    sin_head_kernel<<<B, 64, 0, stream>>>(POOL, lin1w, lin1b, lin2w, lin2b,
                                          (float*)d_out);
}
